// IBRNet_Aggregator_81406810129172
// MI455X (gfx1250) — matrix operand order repaired and checked by reading, not compiled, not run on hardware
//
#include <hip/hip_runtime.h>
#include <hip/hip_bf16.h>

// ---------------------------------------------------------------------------
// IBRNet aggregator on gfx1250 (CDNA5): all pointwise convs as
// v_wmma_f32_16x16x32_f16 over LDS-resident activation tiles.
// Block = 256 threads (8 x wave32). Each block owns 16 voxels x 8 views
// (=128 matmul rows) and runs the full layer stack in LDS.
// ---------------------------------------------------------------------------

#define KSA 96   // actA row stride (halves) : holds up to 96 channels (x96 / gf)
#define KSB 64   // actB row stride (halves) : holds up to 64 channels
#define WKS 96   // staged-weight row stride (halves), [out_ch][k]

typedef __attribute__((ext_vector_type(16))) _Float16 v16h;
typedef __attribute__((ext_vector_type(8)))  float    v8f;

union F16Frag { v16h v; float4 q[2]; };

__device__ __forceinline__ float eluf(float x)  { return x > 0.f ? x : __expf(x) - 1.f; }
__device__ __forceinline__ float sigmf(float x) { return 1.f / (1.f + __expf(-x)); }
__device__ __forceinline__ v8f   zero8()        { v8f c = {0.f,0.f,0.f,0.f,0.f,0.f,0.f,0.f}; return c; }

// One 16x16 C-tile, accumulating over `kchunks` K-slices of 32.
// A: 16 rows (rowBase..rowBase+15) of `act` (f16, stride aks), column offset
//    baked into `act` pointer (must be a multiple of 32 halves for alignment).
// B: staged weights Wt[n][k] (f16, stride WKS), N-tile `ntile`.
// Contiguous K order (k = 16*g + s) on A and B alike, not the ISA table order (wave32).
__device__ __forceinline__ v8f mm_tile(const _Float16* __restrict__ act, int aks,
                                       int rowBase, const _Float16* __restrict__ wt,
                                       int ntile, int kchunks, int lane, v8f c) {
  const int n = lane & 15;
  const int g = lane >> 4;
  const _Float16* arow = act + (size_t)(rowBase + n) * aks + g * 16;  // A row = lane&15
  const _Float16* brow = wt + (size_t)(ntile * 16 + n) * WKS + g * 16;
#pragma unroll
  for (int kc = 0; kc < kchunks; ++kc) {
    F16Frag a, b;
    const float4* ap = reinterpret_cast<const float4*>(arow + kc * 32);
    a.q[0] = ap[0];            // K = kc*32 + g*16 + (0..7)
    a.q[1] = ap[1];            // K = kc*32 + g*16 + (8..15)
    const float4* bp = reinterpret_cast<const float4*>(brow + kc * 32);
    b.q[0] = bp[0];            // K = kc*32 + g*16 + (0..7)
    b.q[1] = bp[1];            // K = kc*32 + g*16 + (8..15)
    c = __builtin_amdgcn_wmma_f32_16x16x32_f16(false, a.v, false, b.v,
                                               (short)0, c, false, false);
  }
  return c;
}

// C-tile store with bias + ELU into an LDS activation buffer.
__device__ __forceinline__ void store_elu(v8f c, _Float16* act, int aks, int rowBase,
                                          int colBase, int ntile, const float* bias,
                                          int lane) {
  const int n = lane & 15, g = lane >> 4;
  const float bv = bias[ntile * 16 + n];
#pragma unroll
  for (int r = 0; r < 8; ++r) {
    const int row = rowBase + 8 * g + r;      // C: m = 8*g + r, n = lane&15
    act[(size_t)row * aks + colBase + ntile * 16 + n] = (_Float16)eluf(c[r] + bv);
  }
}

// Stage an OxI f32 weight matrix into LDS as f16 [n][k], zero-padded to 64x96.
__device__ __forceinline__ void stage_w(const float* __restrict__ w,
                                        const float* __restrict__ b,
                                        int Cout, int Cin,
                                        _Float16* Wt, float* biasS, int tid) {
  for (int i = tid; i < 64 * 96; i += 256) {
    const int nn = i / 96, k = i % 96;
    float v = (nn < Cout && k < Cin) ? w[nn * Cin + k] : 0.f;
    Wt[nn * WKS + k] = (_Float16)v;
  }
  for (int i = tid; i < 64; i += 256) biasS[i] = (i < Cout) ? b[i] : 0.f;
}

__global__ __launch_bounds__(256) void ibr_agg_kernel(
    const float* __restrict__ features, const float* __restrict__ mask,
    const float* __restrict__ depth,    const float* __restrict__ dir,
    const float* rde_w1,  const float* rde_b1,  const float* rde_w2,  const float* rde_b2,
    const float* base_w1, const float* base_b1, const float* base_w2, const float* base_b2,
    const float* vis_w1,  const float* vis_b1,  const float* vis_w2,  const float* vis_b2,
    const float* vis2_w1, const float* vis2_b1, const float* vis2_w2, const float* vis2_b2,
    const float* stat_w,  const float* stat_b,  float* __restrict__ out) {

  __shared__ __align__(16) _Float16 actA[128 * KSA];  // 24.0 KB
  __shared__ __align__(16) _Float16 actB[128 * KSB];  // 16.0 KB
  __shared__ __align__(16) _Float16 Wt[64 * WKS];     // 12.0 KB
  __shared__ float biasS[64];
  __shared__ float maskS[128], weightS[128], visS[128], vis2S[128], weight2S[128];
  __shared__ float wsumS[16], wbarS[16];
  __shared__ float meanS[512], varS[512];

  const int    tid     = threadIdx.x;
  const int    lane    = tid & 31;
  const int    w       = tid >> 5;                 // wave id = view id
  const int    bb      = blockIdx.x >> 11;         // batch
  const int    p0      = (blockIdx.x & 2047) << 4; // first voxel of this tile
  const size_t P       = 32768;                    // G^3
  const int    rowBase = w * 16;

  // Warm L2/GL1 for the features tile (global_prefetch_b8).
  {
    const float* fbase = features + (size_t)bb * 256 * P + p0;
    for (int i = tid; i < 256; i += 256) __builtin_prefetch(fbase + (size_t)i * P, 0, 0);
  }

  // ---- 0) zero activation buffers (K-padding safety) ----------------------
  for (int i = tid; i < 128 * KSA; i += 256) actA[i] = (_Float16)0.f;
  for (int i = tid; i < 128 * KSB; i += 256) actB[i] = (_Float16)0.f;
  __syncthreads();

  // ---- 1) load rde input [dir(3) | depth(1)] and mask ---------------------
  for (int i = tid; i < 512; i += 256) {
    const int row = i >> 2, c = i & 3, v = row >> 4, q = row & 15;
    const float val = (c < 3) ? dir[(((size_t)bb * 8 + v) * 3 + c) * P + p0 + q]
                              : depth[((size_t)bb * 8 + v) * P + p0 + q];
    actB[row * KSB + c] = (_Float16)val;
  }
  for (int i = tid; i < 128; i += 256)
    maskS[i] = mask[((size_t)bb * 8 + (i >> 4)) * P + p0 + (i & 15)];
  __syncthreads();

  // ---- 2) rde layer 1: 4 -> 16 (K padded to 32) ---------------------------
  stage_w(rde_w1, rde_b1, 16, 4, Wt, biasS, tid);
  __syncthreads();
  { v8f c = mm_tile(actB, KSB, rowBase, Wt, 0, 1, lane, zero8());
    store_elu(c, actA, KSA, rowBase, 0, 0, biasS, lane); }
  __syncthreads();

  // ---- 3) rde layer 2: 16 -> 32 ------------------------------------------
  stage_w(rde_w2, rde_b2, 32, 16, Wt, biasS, tid);
  __syncthreads();
#pragma unroll
  for (int nt = 0; nt < 2; ++nt) {
    v8f c = mm_tile(actA, KSA, rowBase, Wt, nt, 1, lane, zero8());
    store_elu(c, actA, KSA, rowBase, 64, nt, biasS, lane);  // rde -> cols 64..95
  }
  __syncthreads();

  // ---- 4) feats = features + rde; weight = mask / (sum_N mask + eps) ------
  for (int i = tid; i < 4096; i += 256) {
    const int row = i >> 5, c = i & 31, v = row >> 4, q = row & 15;
    const float f = features[(((size_t)bb * 8 + v) * 32 + c) * P + p0 + q];
    const int idx = row * KSA + 64 + c;
    actA[idx] = (_Float16)((float)actA[idx] + f);
  }
  __syncthreads();
  if (tid < 16) { float s = 0.f; for (int v = 0; v < 8; ++v) s += maskS[v * 16 + tid]; wsumS[tid] = s; }
  __syncthreads();
  if (tid < 128) weightS[tid] = maskS[tid] / (wsumS[tid & 15] + 1e-8f);
  __syncthreads();

  // ---- 5) weighted mean/var over views; broadcast gf into cols 0..63 -----
  for (int idx = tid; idx < 512; idx += 256) {
    const int q = idx >> 5, c = idx & 31;
    float mu = 0.f;
    for (int v = 0; v < 8; ++v) mu += weightS[v * 16 + q] * (float)actA[(v * 16 + q) * KSA + 64 + c];
    float va = 0.f;
    for (int v = 0; v < 8; ++v) {
      const float d = (float)actA[(v * 16 + q) * KSA + 64 + c] - mu;
      va += weightS[v * 16 + q] * d * d;
    }
    meanS[idx] = mu; varS[idx] = va;
  }
  __syncthreads();
  for (int i = tid; i < 8192; i += 256) {
    const int row = i >> 6, c = i & 63, q = row & 15;
    actA[row * KSA + c] = (_Float16)((c < 32) ? meanS[(q << 5) + c] : varS[(q << 5) + (c - 32)]);
  }
  __syncthreads();

  // ---- 6) base layer 1: 96 -> 64 ------------------------------------------
  stage_w(base_w1, base_b1, 64, 96, Wt, biasS, tid);
  __syncthreads();
#pragma unroll
  for (int nt = 0; nt < 4; ++nt) {
    v8f c = mm_tile(actA, KSA, rowBase, Wt, nt, 3, lane, zero8());
    store_elu(c, actB, KSB, rowBase, 0, nt, biasS, lane);
  }
  __syncthreads();

  // ---- 7) base layer 2: 64 -> 32  (x -> actA cols 0..31) ------------------
  stage_w(base_w2, base_b2, 32, 64, Wt, biasS, tid);
  __syncthreads();
#pragma unroll
  for (int nt = 0; nt < 2; ++nt) {
    v8f c = mm_tile(actB, KSB, rowBase, Wt, nt, 2, lane, zero8());
    store_elu(c, actA, KSA, rowBase, 0, nt, biasS, lane);
  }
  __syncthreads();

  // ---- 8) xw = x * weight -> actB cols 0..31 ------------------------------
  for (int i = tid; i < 4096; i += 256) {
    const int row = i >> 5, c = i & 31;
    actB[row * KSB + c] = (_Float16)((float)actA[row * KSA + c] * weightS[row]);
  }
  __syncthreads();

  // ---- 9) vis layer 1: 32 -> 32 -> actB cols 32..63 -----------------------
  stage_w(vis_w1, vis_b1, 32, 32, Wt, biasS, tid);
  __syncthreads();
#pragma unroll
  for (int nt = 0; nt < 2; ++nt) {
    v8f c = mm_tile(actB, KSB, rowBase, Wt, nt, 1, lane, zero8());
    store_elu(c, actB, KSB, rowBase, 32, nt, biasS, lane);
  }
  __syncthreads();

  // ---- 10) vis layer 2: 32 -> 33; fuse x += x_res, extract vis channel ----
  stage_w(vis_w2, vis_b2, 33, 32, Wt, biasS, tid);
  __syncthreads();
  {
    const int n = lane & 15, g = lane >> 4;
#pragma unroll
    for (int nt = 0; nt < 2; ++nt) {               // x_res channels 0..31
      v8f c = mm_tile(actB + 32, KSB, rowBase, Wt, nt, 1, lane, zero8());
      const float bv = biasS[nt * 16 + n];
#pragma unroll
      for (int r = 0; r < 8; ++r) {
        const int row = rowBase + 8 * g + r;
        const int idx = row * KSA + nt * 16 + n;
        actA[idx] = (_Float16)((float)actA[idx] + eluf(c[r] + bv));  // x += elu(x_res)
      }
    }
    { v8f c = mm_tile(actB + 32, KSB, rowBase, Wt, 2, 1, lane, zero8());  // channel 32
      if (n == 0) {
        const float bv = biasS[32];
#pragma unroll
        for (int r = 0; r < 8; ++r) {
          const int row = rowBase + 8 * g + r;
          visS[row] = sigmf(eluf(c[r] + bv)) * maskS[row];
        }
      }
    }
  }
  __syncthreads();

  // ---- 11) xv2 = x * vis -> actB cols 0..31 -------------------------------
  for (int i = tid; i < 4096; i += 256) {
    const int row = i >> 5, c = i & 31;
    actB[row * KSB + c] = (_Float16)((float)actA[row * KSA + c] * visS[row]);
  }
  __syncthreads();

  // ---- 12) vis2 layer 1: 32 -> 32 -> actB cols 32..63 ---------------------
  stage_w(vis2_w1, vis2_b1, 32, 32, Wt, biasS, tid);
  __syncthreads();
#pragma unroll
  for (int nt = 0; nt < 2; ++nt) {
    v8f c = mm_tile(actB, KSB, rowBase, Wt, nt, 1, lane, zero8());
    store_elu(c, actB, KSB, rowBase, 32, nt, biasS, lane);
  }
  __syncthreads();

  // ---- 13) vis2 layer 2: 32 -> 1; sigmoid * mask --------------------------
  stage_w(vis2_w2, vis2_b2, 1, 32, Wt, biasS, tid);
  __syncthreads();
  {
    v8f c = mm_tile(actB + 32, KSB, rowBase, Wt, 0, 1, lane, zero8());
    const int n = lane & 15, g = lane >> 4;
    if (n == 0) {
      const float bv = biasS[0];
#pragma unroll
      for (int r = 0; r < 8; ++r) {
        const int row = rowBase + 8 * g + r;
        vis2S[row] = sigmf(c[r] + bv) * maskS[row];
      }
    }
  }
  __syncthreads();

  // ---- 14) weight2 = vis2/(sum+eps); wbar = mean_N(weight2) ---------------
  if (tid < 16) {
    float s = 0.f; for (int v = 0; v < 8; ++v) s += vis2S[v * 16 + tid];
    wsumS[tid] = s;
    wbarS[tid] = (s / (s + 1e-8f)) * 0.125f;
  }
  __syncthreads();
  if (tid < 128) weight2S[tid] = vis2S[tid] / (wsumS[tid & 15] + 1e-8f);
  __syncthreads();

  // ---- 15) mean2/var2 over x with weight2 ---------------------------------
  for (int idx = tid; idx < 512; idx += 256) {
    const int q = idx >> 5, c = idx & 31;
    float mu = 0.f;
    for (int v = 0; v < 8; ++v) mu += weight2S[v * 16 + q] * (float)actA[(v * 16 + q) * KSA + c];
    float va = 0.f;
    for (int v = 0; v < 8; ++v) {
      const float d = (float)actA[(v * 16 + q) * KSA + c] - mu;
      va += weight2S[v * 16 + q] * d * d;
    }
    meanS[idx] = mu; varS[idx] = va;
  }
  __syncthreads();

  // ---- 16) gf = [mean2 | var2 | wbar | 0-pad] in actA rows 0..15 ----------
  for (int i = tid; i < 16 * 96; i += 256) {
    const int row = i / 96, c = i % 96;
    float val;
    if (c < 32)       val = meanS[(row << 5) + c];
    else if (c < 64)  val = varS[(row << 5) + (c - 32)];
    else if (c == 64) val = wbarS[row];
    else              val = 0.f;
    actA[row * KSA + c] = (_Float16)val;
  }
  __syncthreads();

  // ---- 17) stat: 65 -> 32, ELU, store to global ---------------------------
  stage_w(stat_w, stat_b, 32, 65, Wt, biasS, tid);
  __syncthreads();
  if (w < 2) {                                   // waves 0,1: out channels 0..31
    v8f c = mm_tile(actA, KSA, 0, Wt, w, 3, lane, zero8());
    const int n = lane & 15, g = lane >> 4;
    const float bv = biasS[w * 16 + n];
#pragma unroll
    for (int r = 0; r < 8; ++r) {
      const int q = 8 * g + r;                   // voxel within tile
      out[((size_t)bb * 32 + w * 16 + n) * P + p0 + q] = eluf(c[r] + bv);
    }
  }
}

extern "C" void kernel_launch(void* const* d_in, const int* in_sizes, int n_in,
                              void* d_out, int out_size, void* d_ws, size_t ws_size,
                              hipStream_t stream) {
  (void)in_sizes; (void)n_in; (void)d_ws; (void)ws_size; (void)out_size;
  const float* f[22];
  for (int i = 0; i < 22; ++i) f[i] = (const float*)d_in[i];
  // grid: B(2) * (32768/16) voxel tiles = 4096 blocks, 256 threads (8 wave32)
  ibr_agg_kernel<<<dim3(4096), dim3(256), 0, stream>>>(
      f[0], f[1], f[2], f[3], f[4], f[5], f[6], f[7], f[8], f[9], f[10], f[11],
      f[12], f[13], f[14], f[15], f[16], f[17], f[18], f[19], f[20], f[21],
      (float*)d_out);
}
